// GIN_classifier_to_explain_v2_54322746360002
// MI455X (gfx1250) — compile-verified, matrix operand order checked by reading, not run on hardware
//
#include <hip/hip_runtime.h>
#include <math.h>

#define SLOPE 0.01f

typedef __attribute__((ext_vector_type(16))) __bf16 v16bf;
typedef __attribute__((ext_vector_type(8)))  float  v8f;

__device__ __forceinline__ float lrelu(float v) { return v >= 0.f ? v : SLOPE * v; }

// ---------------------------------------------------------------------------
// K1: p[N,8] = x[N,128] @ W1a[128,8]   via v_wmma_f32_16x16x32_bf16
// One wave per 16-row tile; K = 128 -> 4 WMMA steps. W1a zero-padded to 16 cols.
// f32 -> bf16 via native casts (compiler emits packed hardware converts).
// ---------------------------------------------------------------------------
__global__ void k1_gemm_x_w1a(const float* __restrict__ x,
                              const float* __restrict__ W1a,
                              float* __restrict__ p, int N, int tiles) {
  const int lane = threadIdx.x & 31;
  const int wave = threadIdx.x >> 5;
  const int wavesPerBlock = blockDim.x >> 5;
  const int gwave  = blockIdx.x * wavesPerBlock + wave;
  const int nWaves = gridDim.x * wavesPerBlock;

  const int n    = lane & 15;        // A: row-in-tile  /  B,D: column
  const int hi   = (lane >> 4) & 1;  // half-wave selects K sub-range
  const int base = hi * 8;

  // --- Build B (W1a -> bf16, padded to 16 cols) once per wave: 4 K-chunks ---
  // 16-bit B layout (ISA 7.12.2, mirror of A with lane = N column):
  // element j maps to krel = j + (j&8) + base
  v16bf bch[4];
#pragma unroll
  for (int c = 0; c < 4; ++c) {
    const int kb = c * 32;
    v16bf b;
#pragma unroll
    for (int j = 0; j < 16; ++j) {
      const int k = kb + j + (j & 8) + base;
      const float f = (n < 8) ? W1a[k * 8 + n] : 0.f;
      b[j] = (__bf16)f;
    }
    bch[c] = b;
  }

  for (int t = gwave; t < tiles; t += nWaves) {
    const int row0   = t * 16;
    const int row    = row0 + n;
    const int rclamp = (row < N) ? row : 0;
    v8f acc = {};
#pragma unroll
    for (int c = 0; c < 4; ++c) {
      const int kb = c * 32;
      // A layout: elements 0..7 = K kb+base..+7 ; elements 8..15 = +16..+23
      const float* rp = x + (size_t)rclamp * 128 + kb + base;
      const float4 a0 = *(const float4*)(rp + 0);
      const float4 a1 = *(const float4*)(rp + 4);
      const float4 a2 = *(const float4*)(rp + 16);
      const float4 a3 = *(const float4*)(rp + 20);
      v16bf a;
      a[0]  = (__bf16)a0.x; a[1]  = (__bf16)a0.y;
      a[2]  = (__bf16)a0.z; a[3]  = (__bf16)a0.w;
      a[4]  = (__bf16)a1.x; a[5]  = (__bf16)a1.y;
      a[6]  = (__bf16)a1.z; a[7]  = (__bf16)a1.w;
      a[8]  = (__bf16)a2.x; a[9]  = (__bf16)a2.y;
      a[10] = (__bf16)a2.z; a[11] = (__bf16)a2.w;
      a[12] = (__bf16)a3.x; a[13] = (__bf16)a3.y;
      a[14] = (__bf16)a3.z; a[15] = (__bf16)a3.w;
      // (neg_a, A, neg_b, B, c_mod, C, reuse_a, reuse_b)
      acc = __builtin_amdgcn_wmma_f32_16x16x32_bf16(false, a, false, bch[c],
                                                    (short)0, acc, false, false);
    }
    // D layout: lane col n=lane&15; VGPR v -> row v (lanes 0-15) / v+8 (16-31)
    if (n < 8) {
#pragma unroll
      for (int v = 0; v < 8; ++v) {
        const int r = row0 + v + base;
        if (r < N) p[(size_t)r * 8 + n] = acc[v];
      }
    }
  }
}

// ---------------------------------------------------------------------------
// K2: q[dst] += p[src] (8-wide), one thread per edge.
// ---------------------------------------------------------------------------
__global__ void k2_scatter(const int* __restrict__ src, const int* __restrict__ dst,
                           const float* __restrict__ p, float* __restrict__ q, int E) {
  const int e = blockIdx.x * blockDim.x + threadIdx.x;
  if (e >= E) return;
  const int s = src[e], d = dst[e];
  const float4* ps = (const float4*)(p + (size_t)s * 8);
  const float4 a = ps[0], b = ps[1];
  float* qd = q + (size_t)d * 8;
  atomicAdd(qd + 0, a.x); atomicAdd(qd + 1, a.y);
  atomicAdd(qd + 2, a.z); atomicAdd(qd + 3, a.w);
  atomicAdd(qd + 4, b.x); atomicAdd(qd + 5, b.y);
  atomicAdd(qd + 6, b.z); atomicAdd(qd + 7, b.w);
}

// ---------------------------------------------------------------------------
// K3: z = lrelu(p+q+b1a); h = lrelu(z@W1b+b1b); p <- h@W2a; q <- 0
// ---------------------------------------------------------------------------
__global__ void k3_node1(float* __restrict__ p, float* __restrict__ q,
                         const float* __restrict__ b1a, const float* __restrict__ W1b,
                         const float* __restrict__ b1b, const float* __restrict__ W2a,
                         int N) {
  const int i = blockIdx.x * blockDim.x + threadIdx.x;
  if (i >= N) return;
  float z[8], h[8], p2[8];
#pragma unroll
  for (int c = 0; c < 8; ++c)
    z[c] = lrelu(p[(size_t)i * 8 + c] + q[(size_t)i * 8 + c] + b1a[c]);
#pragma unroll
  for (int o = 0; o < 8; ++o) {
    float acc = b1b[o];
#pragma unroll
    for (int k = 0; k < 8; ++k) acc += z[k] * W1b[k * 8 + o];
    h[o] = lrelu(acc);                    // conv1 output + inter-layer lrelu
  }
#pragma unroll
  for (int o = 0; o < 8; ++o) {
    float acc = 0.f;
#pragma unroll
    for (int k = 0; k < 8; ++k) acc += h[k] * W2a[k * 8 + o];
    p2[o] = acc;                          // bias b2a added post-aggregation
  }
#pragma unroll
  for (int c = 0; c < 8; ++c) {
    p[(size_t)i * 8 + c] = p2[c];
    q[(size_t)i * 8 + c] = 0.f;           // re-zero q for second scatter
  }
}

// ---------------------------------------------------------------------------
// K5: z2 = lrelu(p2+q2+b2a); h2 = z2@W2b+b2b; y = lrelu(h2)@Wf1 + bf1
// ---------------------------------------------------------------------------
__global__ void k5_node2(const float* __restrict__ p, const float* __restrict__ q,
                         const float* __restrict__ b2a, const float* __restrict__ W2b,
                         const float* __restrict__ b2b, const float* __restrict__ Wf1,
                         const float* __restrict__ bf1, float* __restrict__ y, int N) {
  const int i = blockIdx.x * blockDim.x + threadIdx.x;
  if (i >= N) return;
  float z[8];
#pragma unroll
  for (int c = 0; c < 8; ++c)
    z[c] = lrelu(p[(size_t)i * 8 + c] + q[(size_t)i * 8 + c] + b2a[c]);
  float acc = bf1[0];
#pragma unroll
  for (int o = 0; o < 8; ++o) {
    float h = b2b[o];
#pragma unroll
    for (int k = 0; k < 8; ++k) h += z[k] * W2b[k * 8 + o];
    acc += lrelu(h) * Wf1[o];
  }
  y[i] = acc;
}

// ---------------------------------------------------------------------------
// K6: per-graph FC2 over nNodes + 2-class log_softmax. One block per graph.
// ---------------------------------------------------------------------------
__global__ void k6_graph(const float* __restrict__ y, const float* __restrict__ Wf2,
                         const float* __restrict__ bf2, float* __restrict__ out,
                         int nNodes) {
  __shared__ float s0[256], s1[256];
  const int g = blockIdx.x;
  float a0 = 0.f, a1 = 0.f;
  for (int n = threadIdx.x; n < nNodes; n += blockDim.x) {
    const float v = lrelu(y[(size_t)g * nNodes + n]);
    a0 += v * Wf2[n * 2 + 0];
    a1 += v * Wf2[n * 2 + 1];
  }
  s0[threadIdx.x] = a0; s1[threadIdx.x] = a1;
  __syncthreads();
  for (int off = blockDim.x >> 1; off > 0; off >>= 1) {
    if ((int)threadIdx.x < off) {
      s0[threadIdx.x] += s0[threadIdx.x + off];
      s1[threadIdx.x] += s1[threadIdx.x + off];
    }
    __syncthreads();
  }
  if (threadIdx.x == 0) {
    const float l0 = s0[0] + bf2[0], l1 = s1[0] + bf2[1];
    const float m  = fmaxf(l0, l1);
    const float lse = m + logf(expf(l0 - m) + expf(l1 - m));
    out[g * 2 + 0] = l0 - lse;
    out[g * 2 + 1] = l1 - lse;
  }
}

extern "C" void kernel_launch(void* const* d_in, const int* in_sizes, int n_in,
                              void* d_out, int out_size, void* d_ws, size_t ws_size,
                              hipStream_t stream) {
  const float* x   = (const float*)d_in[0];
  const int*   ei  = (const int*)  d_in[1];
  // d_in[2] = batch (unused: nodes are contiguous per graph)
  const float* W1a = (const float*)d_in[3];
  const float* b1a = (const float*)d_in[4];
  const float* W1b = (const float*)d_in[5];
  const float* b1b = (const float*)d_in[6];
  const float* W2a = (const float*)d_in[7];
  const float* b2a = (const float*)d_in[8];
  const float* W2b = (const float*)d_in[9];
  const float* b2b = (const float*)d_in[10];
  const float* Wf1 = (const float*)d_in[11];
  const float* bf1 = (const float*)d_in[12];
  const float* Wf2 = (const float*)d_in[13];
  const float* bf2 = (const float*)d_in[14];

  const int N      = in_sizes[0] / 128;   // 128000
  const int E      = in_sizes[1] / 2;     // 2048000
  const int B      = out_size / 2;        // 64
  const int nNodes = N / B;               // 2000

  float* bufP = (float*)d_ws;             // [N,8]
  float* bufQ = bufP + (size_t)N * 8;     // [N,8]
  float* bufY = bufQ + (size_t)N * 8;     // [N]

  const int* srcI = ei;
  const int* dstI = ei + E;

  hipMemsetAsync(bufQ, 0, (size_t)N * 8 * sizeof(float), stream);

  const int tiles = (N + 15) / 16;
  k1_gemm_x_w1a<<<250, 256, 0, stream>>>(x, W1a, bufP, N, tiles);

  const int nBlkE = (E + 255) / 256;
  const int nBlkN = (N + 255) / 256;

  k2_scatter<<<nBlkE, 256, 0, stream>>>(srcI, dstI, bufP, bufQ, E);
  k3_node1 <<<nBlkN, 256, 0, stream>>>(bufP, bufQ, b1a, W1b, b1b, W2a, N);
  k2_scatter<<<nBlkE, 256, 0, stream>>>(srcI, dstI, bufP, bufQ, E);
  k5_node2 <<<nBlkN, 256, 0, stream>>>(bufP, bufQ, b2a, W2b, b2b, Wf1, bf1, bufY, N);
  k6_graph <<<B, 256, 0, stream>>>(bufY, Wf2, bf2, (float*)d_out, nNodes);
}
